// INP_Former_60662118088917
// MI455X (gfx1250) — matrix operand order repaired and checked by reading, not compiled, not run on hardware
//
#include <hip/hip_runtime.h>

typedef __attribute__((ext_vector_type(16))) __bf16 v16bf;
typedef __attribute__((ext_vector_type(8)))  float  v8f;
typedef __attribute__((ext_vector_type(4)))  float  v4f;

namespace {
constexpr int   kB   = 64;
constexpr int   kL   = 1370;
constexpr int   kK   = 64;
constexpr int   kD   = 768;
constexpr float kEps = 1e-8f;

constexpr int kDSteps       = kD / 32;                         // 24 WMMA K-steps
constexpr int kNTiles       = kK / 16;                         // 4 tiles of 16 key columns
constexpr int kPackPerBatch = kDSteps * kNTiles * 32 * 16;     // 49152 bf16 per batch
constexpr int kChunkSteps   = kDSteps / 2;                     // 12 steps per LDS chunk
constexpr int kChunkElems   = kChunkSteps * kNTiles * 32 * 16; // 24576 bf16 = 48 KB
constexpr int kLTile        = 128;                             // L rows per block (8 waves x 16)
constexpr int kLBlocks      = (kL + kLTile - 1) / kLTile;      // 11
}

// ---------------------------------------------------------------------------
// Phase 1: normalize key rows (f32 math) and scatter normalized bf16 values
// into the exact WMMA B-fragment layout:
//   ws[b][ (((d/32)*4 + k/16)*32 + k%16 + 16*((d>>3)&1))*16 + (d&7) + 8*((d>>4)&1) ]  <-  kn[k][d]
// so phase 2's LDS staging is a linear vector copy and B-fragment loads are
// 32 contiguous bytes per lane (conflict-free).
// ---------------------------------------------------------------------------
__global__ __launch_bounds__(256) void normalize_pack_keys(
    const float* __restrict__ keys, __bf16* __restrict__ wsk) {
  const int wave = threadIdx.x >> 5;
  const int lane = threadIdx.x & 31;
  const int g = blockIdx.x * 8 + wave;   // global key row, g = b*64 + k
  const int b = g >> 6;
  const int k = g & 63;

  const float* row = keys + (size_t)g * kD;
  float vals[24];
  float ss = 0.0f;
#pragma unroll
  for (int i = 0; i < 24; ++i) {
    float x = row[lane + 32 * i];
    vals[i] = x;
    ss = fmaf(x, x, ss);
  }
#pragma unroll
  for (int off = 16; off >= 1; off >>= 1) ss += __shfl_xor(ss, off, 32);
  const float rn = 1.0f / fmaxf(sqrtf(ss), kEps);

  __bf16* dst = wsk + (size_t)b * kPackPerBatch;
  const int kt = k >> 4;
#pragma unroll
  for (int i = 0; i < 24; ++i) {
    // element d = 32*i + lane  ->  packed index ((i*4 + kt)*32 + (k&15) + 16*((lane>>3)&1))*16 + (lane&7) + 8*(lane>>4)
    dst[((i * kNTiles + kt) * 32 + (k & 15) + ((lane >> 3) & 1) * 16) * 16 + (lane & 7) + ((lane >> 4) << 3)] = (__bf16)(vals[i] * rn);
  }
}

// ---------------------------------------------------------------------------
// Phase 2 helpers
// ---------------------------------------------------------------------------
__device__ inline v4f ntload4(const float* p) {
  // q is 269 MB streamed exactly once: non-temporal so it doesn't evict the
  // 11x-reused key tiles from the 192 MB L2.
  return __builtin_nontemporal_load((const v4f*)p);
}
__device__ inline void cvt4(v16bf& a, int base, v4f f) {
  a[base + 0] = (__bf16)f.x;
  a[base + 1] = (__bf16)f.y;
  a[base + 2] = (__bf16)f.z;
  a[base + 3] = (__bf16)f.w;
}
__device__ inline float ssq4(float acc, v4f f) {
  acc = fmaf(f.x, f.x, acc);
  acc = fmaf(f.y, f.y, acc);
  acc = fmaf(f.z, f.z, acc);
  acc = fmaf(f.w, f.w, acc);
  return acc;
}

// ---------------------------------------------------------------------------
// Phase 2: per block = (batch, 128 L-rows). Each wave owns a 16(L) x 64(K)
// output tile: D-loop of v_wmma_f32_16x16x32_bf16. Loop body order is chosen
// to minimize live ranges (no accumulator copies / hazard nops):
//   1) convert current A floats (frees them)   2) prefetch next A (global)
//   3) batch 8x ds_load_b128 for B             4) 4x WMMA consuming B at once
// #pragma unroll 1 keeps register pressure low enough that the accumulators
// stay pinned in place across the whole loop.
// ---------------------------------------------------------------------------
__global__ __launch_bounds__(256) void gather_loss_main(
    const float* __restrict__ q, const __bf16* __restrict__ wsk,
    float* __restrict__ partials) {
  __shared__ alignas(32) __bf16 sk[kChunkElems];  // 48 KB key chunk
  __shared__ float s_norm[8][16];
  __shared__ float s_max[8][16];
  __shared__ float s_part[8];

  const int tid  = threadIdx.x;
  const int wave = tid >> 5;
  const int lane = tid & 31;
  const int b    = blockIdx.x / kLBlocks;
  const int lt   = blockIdx.x % kLBlocks;
  const int row_base = lt * kLTile + wave * 16;

  // ISA 16-bit A layout: lane holds row m = lane&15; lane<16 covers
  // k in [0,8)+[16,24), lane>=16 covers [8,16)+[24,32) of each 32-step.
  const int m_mine = row_base + (lane & 15);
  const int ml     = (m_mine < kL) ? m_mine : 0;  // clamp OOB rows (zeroed later)
  const float* qrow = q + ((size_t)b * kL + ml) * (size_t)kD;
  const int koff = (lane >> 4) * 8;

  v8f acc0 = {}, acc1 = {}, acc2 = {}, acc3 = {};
  float ssq = 0.0f;

  for (int c = 0; c < 2; ++c) {
    __syncthreads();  // protect previous chunk before overwrite
    {
      const uint4* src =
          (const uint4*)(wsk + (size_t)b * kPackPerBatch + (size_t)c * kChunkElems);
      uint4* dst = (uint4*)sk;
      for (int i = tid; i < kChunkElems / 8; i += 256) dst[i] = src[i];
    }
    __syncthreads();

    const int dbase = c * (kChunkSteps * 32);
    // --- software pipeline: preload A floats for step 0 of this chunk ---
    v4f f0 = ntload4(qrow + dbase + koff);
    v4f f1 = ntload4(qrow + dbase + koff + 4);
    v4f f2 = ntload4(qrow + dbase + 16 + koff);
    v4f f3 = ntload4(qrow + dbase + 16 + koff + 4);

#pragma unroll 1
    for (int s = 0; s < kChunkSteps; ++s) {
      // 1) consume current A floats: ||q||^2 partial + bf16 A fragment
      ssq = ssq4(ssq, f0); ssq = ssq4(ssq, f1);
      ssq = ssq4(ssq, f2); ssq = ssq4(ssq, f3);
      v16bf a;
      cvt4(a, 0, f0); cvt4(a, 4, f1); cvt4(a, 8, f2); cvt4(a, 12, f3);

      // 2) prefetch next step's A floats (hidden behind the WMMA chain)
      if (s + 1 < kChunkSteps) {
        const int dn = dbase + (s + 1) * 32;
        f0 = ntload4(qrow + dn + koff);
        f1 = ntload4(qrow + dn + koff + 4);
        f2 = ntload4(qrow + dn + 16 + koff);
        f3 = ntload4(qrow + dn + 16 + koff + 4);
      }

      // 3) batch all 8 ds_load_b128 for the 4 B fragments (short live range)
      const v16bf* bp = ((const v16bf*)sk) + (s * kNTiles * 32 + lane);
      v16bf b0 = bp[0];
      v16bf b1 = bp[32];
      v16bf b2 = bp[64];
      v16bf b3 = bp[96];

      // 4) WMMA chain consumes B fragments immediately
      acc0 = __builtin_amdgcn_wmma_f32_16x16x32_bf16(false, a, false, b0,
                                                     (short)0, acc0, false, false);
      acc1 = __builtin_amdgcn_wmma_f32_16x16x32_bf16(false, a, false, b1,
                                                     (short)0, acc1, false, false);
      acc2 = __builtin_amdgcn_wmma_f32_16x16x32_bf16(false, a, false, b2,
                                                     (short)0, acc2, false, false);
      acc3 = __builtin_amdgcn_wmma_f32_16x16x32_bf16(false, a, false, b3,
                                                     (short)0, acc3, false, false);
    }
  }

  // Full ||q||^2 per row: lanes l and l^16 cover complementary k subsets.
  const float nsq = ssq + __shfl_xor(ssq, 16, 32);

  // Row-max over the 64 key columns. C layout: VGPR r -> row (r, lanes<16) or
  // (r+8, lanes>=16); lane%16 -> column. Combine tiles, then xor-reduce halves.
  float rmax[8];
#pragma unroll
  for (int r = 0; r < 8; ++r)
    rmax[r] = fmaxf(fmaxf(acc0[r], acc1[r]), fmaxf(acc2[r], acc3[r]));
#pragma unroll
  for (int off = 1; off < 16; off <<= 1) {
#pragma unroll
    for (int r = 0; r < 8; ++r)
      rmax[r] = fmaxf(rmax[r], __shfl_xor(rmax[r], off, 32));
  }

  if (lane < 16) s_norm[wave][lane] = nsq;
  if (lane == 0) {
#pragma unroll
    for (int r = 0; r < 8; ++r) s_max[wave][r] = rmax[r];
  } else if (lane == 16) {
#pragma unroll
    for (int r = 0; r < 8; ++r) s_max[wave][r + 8] = rmax[r];
  }
  // same-wave LDS RAW: compiler inserts s_wait_dscnt

  float dist = 0.0f;
  if (lane < 16) {
    const int mm = row_base + lane;
    if (mm < kL) {
      const float nrm = fmaxf(sqrtf(s_norm[wave][lane]), kEps);
      dist = 1.0f - s_max[wave][lane] / nrm;  // 1 - max cos = min (1 - cos)
    }
  }
#pragma unroll
  for (int off = 16; off >= 1; off >>= 1) dist += __shfl_xor(dist, off, 32);
  if (lane == 0) s_part[wave] = dist;
  __syncthreads();
  if (tid == 0) {
    float t = 0.0f;
#pragma unroll
    for (int w = 0; w < 8; ++w) t += s_part[w];
    partials[blockIdx.x] = t;
  }
}

// ---------------------------------------------------------------------------
// Phase 3: deterministic fixed-order tree reduction of block partials.
// ---------------------------------------------------------------------------
__global__ __launch_bounds__(256) void finalize_kernel(
    const float* __restrict__ partials, float* __restrict__ out) {
  __shared__ float s[256];
  const int tid = threadIdx.x;
  float t = 0.0f;
  for (int i = tid; i < kB * kLBlocks; i += 256) t += partials[i];
  s[tid] = t;
  __syncthreads();
  for (int off = 128; off >= 1; off >>= 1) {
    if (tid < off) s[tid] += s[tid + off];
    __syncthreads();
  }
  if (tid == 0) out[0] = s[0] * (1.0f / ((float)kB * (float)kL));
}

extern "C" void kernel_launch(void* const* d_in, const int* in_sizes, int n_in,
                              void* d_out, int out_size, void* d_ws, size_t ws_size,
                              hipStream_t stream) {
  (void)in_sizes; (void)n_in; (void)out_size; (void)ws_size;
  const float* q    = (const float*)d_in[0];
  const float* keys = (const float*)d_in[1];

  __bf16* wsk = (__bf16*)d_ws;  // packed normalized keys: 6,291,456 bytes
  float* partials =
      (float*)((char*)d_ws + (size_t)kB * kPackPerBatch * sizeof(__bf16));

  normalize_pack_keys<<<(kB * kK) / 8, 256, 0, stream>>>(keys, wsk);
  gather_loss_main<<<kB * kLBlocks, 256, 0, stream>>>(q, wsk, partials);
  finalize_kernel<<<1, 256, 0, stream>>>(partials, (float*)d_out);
}
